// Transformer_214748364835
// MI455X (gfx1250) — compile-verified, matrix operand order checked by reading, not run on hardware
//
#include <hip/hip_runtime.h>
#include <hip/hip_bf16.h>
#include <cstdint>

// ---------------------------------------------------------------------------
// CDNA5 / gfx1250: all GEMM-shaped work (conv-as-GEMM, fused projection,
// FF/attn output projections) on v_wmma_f32_16x16x32_bf16 with f32 accum.
// GEMM tiles double-buffered in LDS; A-tile staged via async global->LDS
// (ASYNCcnt) when the toolchain exposes the gfx1250 builtins.
// ---------------------------------------------------------------------------

typedef __attribute__((ext_vector_type(16))) __bf16 v16bf;
typedef __attribute__((ext_vector_type(8)))  float  v8f;
typedef int v4i __attribute__((vector_size(16)));   // matches builtin proto

#define NELEM   (4 * 1024 * 1024)   // 4 x 1024 x 1024
#define NTOK    4096                // 4 * 1024 tokens
#define FUSEDW  4736
#define FFIN    4096

#if defined(__gfx1250__) && \
    __has_builtin(__builtin_amdgcn_global_load_async_to_lds_b128) && \
    __has_builtin(__builtin_amdgcn_s_wait_asynccnt)
#define GEMM_ASYNC 1
#else
#define GEMM_ASYNC 0
#endif

__device__ __forceinline__ unsigned short f32_to_bf16(float f) {
  union { float f; uint32_t u; } v; v.f = f;
  uint32_t u = v.u;
  uint32_t r = u + 0x7FFFu + ((u >> 16) & 1u);  // round-nearest-even
  return (unsigned short)(r >> 16);
}

__device__ __forceinline__ float gelu_exact(float x) {
  return 0.5f * x * (1.0f + erff(x * 0.70710678118654752f));
}

// ---------------------------------------------------------------------------
// bf16 WMMA GEMM: C[M,N] (f32) = A[M,K] * B[K,N], A/B bf16 row-major.
// convMode != 0 => logical K-row kk maps to B memory row (kk & 1023) with
// column shift ((kk >> 10) - 1), zero-padded outside [0, Nvalid)  (k=3 conv).
// Block tile 128x128, 256 threads = 8 waves, wave tile 32x64 (2x4 WMMAs).
// Double-buffered LDS; A staged via GLOBAL_LOAD_ASYNC_TO_LDS_B128 when
// available, B prefetched into registers and transposed on ds_store.
// ---------------------------------------------------------------------------
__global__ __launch_bounds__(256)
void gemm_bf16_wmma(const unsigned short* __restrict__ Ag, int lda,
                    const unsigned short* __restrict__ Bg, int ldb,
                    float* __restrict__ C, int ldc,
                    int M, int N, int K,
                    int convMode, int Nvalid, int accumulate)
{
  __shared__ unsigned short As[2][128 * 40];   // stride 40: 16B-aligned rows
  __shared__ unsigned short Bs[2][128 * 40];   // transposed: Bs[n][k]

  const int t    = threadIdx.x;
  const int lane = t & 31;
  const int wave = t >> 5;
  const int wm   = wave >> 1;     // 0..3 -> M sub-offset * 32
  const int wn   = wave & 1;      // 0..1 -> N sub-offset * 64
  const int bm   = blockIdx.y * 128;
  const int bn   = blockIdx.x * 128;

  v8f acc[2][4];
  #pragma unroll
  for (int mi = 0; mi < 2; ++mi)
    #pragma unroll
    for (int ni = 0; ni < 4; ++ni)
      #pragma unroll
      for (int r = 0; r < 8; ++r) acc[mi][ni][r] = 0.0f;

  const int aRow  = t >> 1;         // 0..127
  const int aHalf = (t & 1) * 16;   // 0 or 16 (elements)
  const int bK    = t >> 3;         // 0..31
  const int bN0   = (t & 7) * 16;   // 0..112 step 16

  // stage A tile (128x32): 2 x 16B per thread, async -> LDS when possible
  auto issueA = [&](int k0, int buf) {
    unsigned short* src = const_cast<unsigned short*>(
        Ag + (size_t)(bm + aRow) * lda + k0 + aHalf);
    unsigned short* dst = &As[buf][aRow * 40 + aHalf];
#if GEMM_ASYNC
    __builtin_amdgcn_global_load_async_to_lds_b128(
        (__attribute__((address_space(1))) v4i*)src,
        (__attribute__((address_space(3))) v4i*)dst, 0, 0);
    __builtin_amdgcn_global_load_async_to_lds_b128(
        (__attribute__((address_space(1))) v4i*)(src + 8),
        (__attribute__((address_space(3))) v4i*)(dst + 8), 0, 0);
#else
    *(uint4*)dst       = *(const uint4*)src;
    *(uint4*)(dst + 8) = *(const uint4*)(src + 8);
    if (k0 + 32 < K) __builtin_prefetch(src + 32, 0, 1);  // global_prefetch_b8
#endif
  };

  // fetch B tile (32x128) into registers (16 bf16 per thread)
  auto loadB = [&](int k0, unsigned short* tv) {
    int kk = k0 + bK;
    int memrow = convMode ? (kk & 1023) : kk;
    int shift  = convMode ? ((kk >> 10) - 1) : 0;
    const unsigned short* brow = Bg + (size_t)memrow * ldb;
    int colbase = bn + bN0 + shift;
    if (shift == 0 && colbase + 16 <= Nvalid) {
      *(uint4*)&tv[0] = *(const uint4*)(brow + colbase);
      *(uint4*)&tv[8] = *(const uint4*)(brow + colbase + 8);
    } else {
      #pragma unroll
      for (int e = 0; e < 16; ++e) {
        int c = colbase + e;
        tv[e] = (c >= 0 && c < Nvalid) ? brow[c] : (unsigned short)0;
      }
    }
  };

  // transpose-store B registers into Bs[buf][n][k]
  auto storeB = [&](const unsigned short* tv, int buf) {
    #pragma unroll
    for (int e = 0; e < 16; ++e)
      Bs[buf][(bN0 + e) * 40 + bK] = tv[e];
  };

  unsigned short tv[16];
  const int KT = K >> 5;          // K / 32 tiles
  issueA(0, 0);
  loadB(0, tv);

  for (int kt = 0; kt < KT; ++kt) {
    const int cur = kt & 1;
    const int nxt = cur ^ 1;

    storeB(tv, cur);
    if (kt + 1 < KT) {
      issueA((kt + 1) << 5, nxt);     // async into idle buffer
      loadB((kt + 1) << 5, tv);       // next B into registers
#if GEMM_ASYNC
      __builtin_amdgcn_s_wait_asynccnt(2);  // retire tile kt's 2 async loads
#endif
    } else {
#if GEMM_ASYNC
      __builtin_amdgcn_s_wait_asynccnt(0);
#endif
    }
    __syncthreads();   // waits own DScnt, then barrier: tile kt visible

    {
      union Frag { uint4 u4[2]; v16bf v; };
      Frag fa[2], fb[4];
      const int lr = lane & 15;
      const int kh = (lane >> 4) * 8;
      #pragma unroll
      for (int mi = 0; mi < 2; ++mi) {
        int row = wm * 32 + mi * 16 + lr;
        fa[mi].u4[0] = *(const uint4*)&As[cur][row * 40 + kh];
        fa[mi].u4[1] = *(const uint4*)&As[cur][row * 40 + kh + 16];
      }
      #pragma unroll
      for (int ni = 0; ni < 4; ++ni) {
        int col = wn * 64 + ni * 16 + lr;
        fb[ni].u4[0] = *(const uint4*)&Bs[cur][col * 40 + kh];
        fb[ni].u4[1] = *(const uint4*)&Bs[cur][col * 40 + kh + 16];
      }
      #pragma unroll
      for (int mi = 0; mi < 2; ++mi)
        #pragma unroll
        for (int ni = 0; ni < 4; ++ni)
          acc[mi][ni] = __builtin_amdgcn_wmma_f32_16x16x32_bf16(
              false, fa[mi].v, false, fb[ni].v, (short)0, acc[mi][ni],
              false, false);
    }
    __syncthreads();   // all reads of buffer `cur` done before it is reused
  }

  // C store (16x16 f32 layout: vgpr r -> M = r + 8*(lane>=16))
  const int lr2 = lane & 15;
  const int hi8 = (lane >> 4) * 8;
  #pragma unroll
  for (int mi = 0; mi < 2; ++mi) {
    #pragma unroll
    for (int ni = 0; ni < 4; ++ni) {
      int r0 = bm + wm * 32 + mi * 16 + hi8;
      int c0 = bn + wn * 64 + ni * 16 + lr2;
      #pragma unroll
      for (int r = 0; r < 8; ++r) {
        float* p = C + (size_t)(r0 + r) * ldc + c0;
        float vv = acc[mi][ni][r];
        *p = accumulate ? (*p + vv) : vv;
      }
    }
  }
}

// ---------------------------------------------------------------------------
// Elementwise / helper kernels
// ---------------------------------------------------------------------------
__global__ void copy_f32_kernel(float* __restrict__ o, const float* __restrict__ x, int n) {
  for (int i = blockIdx.x * blockDim.x + threadIdx.x; i < n; i += gridDim.x * blockDim.x)
    o[i] = x[i];
}

__global__ void cvt_f32_bf16_kernel(unsigned short* __restrict__ o,
                                    const float* __restrict__ x, int n) {
  for (int i = blockIdx.x * blockDim.x + threadIdx.x; i < n; i += gridDim.x * blockDim.x)
    o[i] = f32_to_bf16(x[i]);
}

// conv weight repack: w[co][ci][k] (f32) -> A[co][k*1024 + ci] (bf16), K=3072
__global__ void pack_convw_kernel(unsigned short* __restrict__ o,
                                  const float* __restrict__ w) {
  const int n = 1024 * 1024 * 3;
  for (int i = blockIdx.x * blockDim.x + threadIdx.x; i < n; i += gridDim.x * blockDim.x) {
    int co = i / 3072; int r = i - co * 3072; int k = r >> 10; int ci = r & 1023;
    o[i] = f32_to_bf16(w[(size_t)co * 3072 + ci * 3 + k]);
  }
}

// conv post-op: v = gelu(in + bias[co]); optionally write f32 and/or bf16
__global__ void bias_gelu_conv_kernel(const float* __restrict__ in,
                                      const float* __restrict__ bias,
                                      float* __restrict__ of,
                                      unsigned short* __restrict__ obf, int n) {
  for (int i = blockIdx.x * blockDim.x + threadIdx.x; i < n; i += gridDim.x * blockDim.x) {
    float v = in[i] + bias[(i >> 10) & 1023];
    v = gelu_exact(v);
    if (of)  of[i] = v;
    if (obf) obf[i] = f32_to_bf16(v);
  }
}

// x = conv_stack_out + 2*x   (inner residual + outer residual)
__global__ void conv_residual_kernel(float* __restrict__ act,
                                     const float* __restrict__ h, int n) {
  for (int i = blockIdx.x * blockDim.x + threadIdx.x; i < n; i += gridDim.x * blockDim.x)
    act[i] = h[i] + 2.0f * act[i];
}

// RMSNorm over feature axis (1024, contiguous); bf16 out for the GEMM A side
__global__ __launch_bounds__(256)
void rmsnorm_kernel(const float* __restrict__ x, const float* __restrict__ gg,
                    unsigned short* __restrict__ obf) {
  __shared__ float red[256];
  const int row = blockIdx.x;
  const int t = threadIdx.x;
  const float* xr = x + (size_t)row * 1024;
  float ss = 0.f;
  for (int c = t; c < 1024; c += 256) { float v = xr[c]; ss += v * v; }
  red[t] = ss; __syncthreads();
  for (int off = 128; off > 0; off >>= 1) { if (t < off) red[t] += red[t + off]; __syncthreads(); }
  float norm  = sqrtf(red[0]) * (1.0f / 32.0f);   // ||x|| * d^-0.5
  float scale = 1.0f / fmaxf(norm, 1e-8f);
  for (int c = t; c < 1024; c += 256)
    obf[(size_t)row * 1024 + c] = f32_to_bf16(xr[c] * scale * gg[c]);
}

// Split fused projection: q (roped + scaled, per-head), k (roped, shared), v.
__global__ void ropeqkv_kernel(const float* __restrict__ proj,
                               const float* __restrict__ fb,
                               float* __restrict__ qf,
                               float* __restrict__ kf,
                               float* __restrict__ vf) {
  const int total = NTOK * 352;
  for (int idx = blockIdx.x * blockDim.x + threadIdx.x; idx < total;
       idx += gridDim.x * blockDim.x) {
    int tok = idx / 352; int j = idx - tok * 352;
    int b = tok >> 10, i = tok & 1023;
    const float* pr = proj + (size_t)tok * FUSEDW;
    if (j < 256) {
      int h = j >> 5, tt = j & 31;
      int c1 = h * 64 + tt, c2 = c1 + 32;
      float v1 = pr[c1] + fb[c1], v2 = pr[c2] + fb[c2];
      float fr = (float)i * __powf(10000.0f, -(float)tt * (1.0f / 32.0f));
      float cs = __cosf(fr), sn = __sinf(fr);
      float* q = qf + ((size_t)(b * 8 + h) * 1024 + i) * 64;
      q[tt]      = (v1 * cs - v2 * sn) * 0.125f;   // * dh^-0.5
      q[tt + 32] = (v2 * cs + v1 * sn) * 0.125f;
    } else if (j < 288) {
      int tt = j - 256;
      int c1 = 512 + tt, c2 = c1 + 32;
      float v1 = pr[c1] + fb[c1], v2 = pr[c2] + fb[c2];
      float fr = (float)i * __powf(10000.0f, -(float)tt * (1.0f / 32.0f));
      float cs = __cosf(fr), sn = __sinf(fr);
      float* k = kf + (size_t)(b * 1024 + i) * 64;
      k[tt]      = v1 * cs - v2 * sn;
      k[tt + 32] = v2 * cs + v1 * sn;
    } else {
      int d = j - 288, c = 576 + d;
      vf[(size_t)(b * 1024 + i) * 64 + d] = pr[c] + fb[c];
    }
  }
}

// ff branch: bf16(gelu(proj[:,640:] + bias)) -> GEMM A operand
__global__ void gelu_ff_kernel(const float* __restrict__ proj,
                               const float* __restrict__ fb,
                               unsigned short* __restrict__ ffbf) {
  const int n = NTOK * FFIN;
  for (int i = blockIdx.x * blockDim.x + threadIdx.x; i < n; i += gridDim.x * blockDim.x) {
    int tok = i >> 12, c = i & 4095;
    float v = proj[(size_t)tok * FUSEDW + 640 + c] + fb[640 + c];
    ffbf[i] = f32_to_bf16(gelu_exact(v));
  }
}

// Causal multi-query attention, one block per (b,h,query row). dh = 64.
__global__ __launch_bounds__(256)
void attn_kernel(const float* __restrict__ qf, const float* __restrict__ kf,
                 const float* __restrict__ vf, unsigned short* __restrict__ aobf) {
  __shared__ float sQ[64];
  __shared__ float sS[1024];
  __shared__ float red[256];
  const int blk = blockIdx.x;
  const int i = blk & 1023, h = (blk >> 10) & 7, b = blk >> 13;
  const int t = threadIdx.x;
  const float* q = qf + ((size_t)(b * 8 + h) * 1024 + i) * 64;
  if (t < 64) sQ[t] = q[t];
  __syncthreads();
  const int nk = i + 1;
  float lmax = -3.0e38f;
  for (int j = t; j < 1024; j += 256) {
    float s = -3.0e38f;
    if (j < nk) {
      const float* kr = kf + (size_t)(b * 1024 + j) * 64;
      float a0 = 0.f;
      #pragma unroll 8
      for (int d = 0; d < 64; ++d) a0 += sQ[d] * kr[d];
      s = a0;
    }
    sS[j] = s;
    lmax = fmaxf(lmax, s);
  }
  red[t] = lmax; __syncthreads();
  for (int off = 128; off > 0; off >>= 1) { if (t < off) red[t] = fmaxf(red[t], red[t + off]); __syncthreads(); }
  const float m = red[0];
  __syncthreads();
  float ls = 0.f;
  for (int j = t; j < 1024; j += 256) {
    float p = (j < nk) ? __expf(sS[j] - m) : 0.0f;
    sS[j] = p;
    ls += p;
  }
  red[t] = ls; __syncthreads();
  for (int off = 128; off > 0; off >>= 1) { if (t < off) red[t] += red[t + off]; __syncthreads(); }
  const float inv = 1.0f / red[0];
  __syncthreads();
  const int d = t & 63, part = t >> 6;
  const int j0 = part * 256;
  const int j1 = (j0 + 256 < nk) ? (j0 + 256) : nk;
  float a0 = 0.f;
  for (int j = j0; j < j1; ++j)
    a0 += sS[j] * vf[(size_t)(b * 1024 + j) * 64 + d];
  red[t] = a0; __syncthreads();
  if (part == 0) {
    float o = (red[d] + red[64 + d] + red[128 + d] + red[192 + d]) * inv;
    aobf[(size_t)(b * 1024 + i) * 512 + h * 64 + d] = f32_to_bf16(o);
  }
}

// x += (c1 + attnout_b) + (c2 + ffout_b)
__global__ void final_add_kernel(float* __restrict__ act, const float* __restrict__ c1,
                                 const float* __restrict__ ab, const float* __restrict__ c2,
                                 const float* __restrict__ fob, int n) {
  for (int i = blockIdx.x * blockDim.x + threadIdx.x; i < n; i += gridDim.x * blockDim.x) {
    int c = i & 1023;
    act[i] += c1[i] + ab[c] + c2[i] + fob[c];
  }
}

// ---------------------------------------------------------------------------
// Host orchestration (graph-capture safe: only kernel launches on `stream`)
// ---------------------------------------------------------------------------
extern "C" void kernel_launch(void* const* d_in, const int* in_sizes, int n_in,
                              void* d_out, int out_size, void* d_ws, size_t ws_size,
                              hipStream_t stream) {
  (void)in_sizes; (void)n_in; (void)out_size; (void)ws_size;

  const float* x_in     = (const float*)d_in[0];
  const float* conv_w   = (const float*)d_in[1];
  const float* conv_b   = (const float*)d_in[2];
  const float* gw       = (const float*)d_in[3];
  const float* fused_w  = (const float*)d_in[4];
  const float* fused_b  = (const float*)d_in[5];
  const float* attout_w = (const float*)d_in[6];
  const float* attout_b = (const float*)d_in[7];
  const float* ffout_w  = (const float*)d_in[8];
  const float* ffout_b  = (const float*)d_in[9];
  float* out = (float*)d_out;

  // workspace layout (~194 MB)
  char* wsp = (char*)d_ws;
  auto alloc = [&](size_t bytes) -> void* {
    void* p = (void*)wsp; wsp += (bytes + 255) & ~(size_t)255; return p;
  };
  float*          act  = (float*)alloc((size_t)NELEM * 4);
  float*          tmpA = (float*)alloc((size_t)NELEM * 4);
  float*          tmpB = (float*)alloc((size_t)NELEM * 4);
  unsigned short* abf  = (unsigned short*)alloc((size_t)NELEM * 2);
  unsigned short* wbf  = (unsigned short*)alloc((size_t)1024 * FUSEDW * 2); // largest B
  float*          proj = (float*)alloc((size_t)NTOK * FUSEDW * 4);
  unsigned short* ffbf = (unsigned short*)alloc((size_t)NTOK * FFIN * 2);
  float*          qf   = (float*)alloc((size_t)NTOK * 512 * 4);
  float*          kf   = (float*)alloc((size_t)4 * 1024 * 64 * 4);
  float*          vf   = (float*)alloc((size_t)4 * 1024 * 64 * 4);
  unsigned short* aobf = (unsigned short*)alloc((size_t)NTOK * 512 * 2);

  const int EWG = 8192, EWB = 256;

  copy_f32_kernel<<<EWG, EWB, 0, stream>>>(act, x_in, NELEM);

  for (int i = 0; i < 2; ++i) {
    const float* cw   = conv_w   + (size_t)i * 2 * 1024 * 1024 * 3;
    const float* cb   = conv_b   + (size_t)i * 2 * 1024;
    const float* g_i  = gw       + (size_t)i * 1024;
    const float* fw_i = fused_w  + (size_t)i * 1024 * FUSEDW;
    const float* fb_i = fused_b  + (size_t)i * FUSEDW;
    const float* aw_i = attout_w + (size_t)i * 512 * 1024;
    const float* ab_i = attout_b + (size_t)i * 1024;
    const float* ow_i = ffout_w  + (size_t)i * 4096 * 1024;
    const float* ob_i = ffout_b  + (size_t)i * 1024;

    // ---------------- conv block (x = conv_stack(x) + 2x) ----------------
    cvt_f32_bf16_kernel<<<EWG, EWB, 0, stream>>>(abf, act, NELEM);
    for (int d = 0; d < 2; ++d) {
      pack_convw_kernel<<<EWG, EWB, 0, stream>>>(wbf, cw + (size_t)d * 1024 * 1024 * 3);
      dim3 cg(1024 / 128, 1024 / 128);
      for (int bb = 0; bb < 4; ++bb)
        gemm_bf16_wmma<<<cg, 256, 0, stream>>>(
            wbf, 3072,
            abf + (size_t)bb * 1024 * 1024, 1024,
            tmpA + (size_t)bb * 1024 * 1024, 1024,
            1024, 1024, 3072, /*convMode=*/1, /*Nvalid=*/1024, /*acc=*/0);
      if (d == 0)
        bias_gelu_conv_kernel<<<EWG, EWB, 0, stream>>>(tmpA, cb, nullptr, abf, NELEM);
      else
        bias_gelu_conv_kernel<<<EWG, EWB, 0, stream>>>(tmpA, cb + 1024, tmpA, nullptr, NELEM);
    }
    conv_residual_kernel<<<EWG, EWB, 0, stream>>>(act, tmpA, NELEM);

    // ---------------- transformer block ----------------
    rmsnorm_kernel<<<NTOK, 256, 0, stream>>>(act, g_i, abf);

    cvt_f32_bf16_kernel<<<EWG, EWB, 0, stream>>>(wbf, fw_i, 1024 * FUSEDW);
    gemm_bf16_wmma<<<dim3(FUSEDW / 128, NTOK / 128), 256, 0, stream>>>(
        abf, 1024, wbf, FUSEDW, proj, FUSEDW,
        NTOK, FUSEDW, 1024, 0, FUSEDW, 0);

    ropeqkv_kernel<<<5632, 256, 0, stream>>>(proj, fb_i, qf, kf, vf);
    gelu_ff_kernel<<<EWG, EWB, 0, stream>>>(proj, fb_i, ffbf);
    attn_kernel<<<32768, 256, 0, stream>>>(qf, kf, vf, aobf);

    cvt_f32_bf16_kernel<<<EWG, EWB, 0, stream>>>(wbf, aw_i, 512 * 1024);
    gemm_bf16_wmma<<<dim3(1024 / 128, NTOK / 128), 256, 0, stream>>>(
        aobf, 512, wbf, 1024, tmpA, 1024,
        NTOK, 1024, 512, 0, 1024, 0);

    cvt_f32_bf16_kernel<<<EWG, EWB, 0, stream>>>(wbf, ow_i, 4096 * 1024);
    gemm_bf16_wmma<<<dim3(1024 / 128, NTOK / 128), 256, 0, stream>>>(
        ffbf, 4096, wbf, 1024, tmpB, 1024,
        NTOK, 1024, 4096, 0, 1024, 0);

    final_add_kernel<<<EWG, EWB, 0, stream>>>(act, tmpA, ab_i, tmpB, ob_i, NELEM);
  }

  copy_f32_kernel<<<EWG, EWB, 0, stream>>>(out, act, NELEM);
}
